// CosinesimilarityAttn_22316650070306
// MI455X (gfx1250) — matrix operand order repaired and checked by reading, not compiled, not run on hardware
//
#include <hip/hip_runtime.h>

typedef _Float16 v4h  __attribute__((ext_vector_type(4)));
typedef _Float16 v8h  __attribute__((ext_vector_type(8)));
typedef _Float16 v16h __attribute__((ext_vector_type(16)));
typedef float    v8f  __attribute__((ext_vector_type(8)));

#define NQ   4096
#define NK   4096
#define DIM  1024
#define EPSN 1e-8f
#define BK1  32    // K per iteration (one 16x16x32 WMMA deep)
#define STR  40    // GEMM1 LDS row stride in halves: 80B rows -> 16B aligned
#define BK2  32
#define LDT  132   // GEMM2 LDS k-row stride in floats: 528B rows -> 16B aligned

// hi/lo error-compensated f16 split: x ~= hi + lo, |x - hi - lo| ~ 2^-22 |x|
struct H4 { v4h h, l; };
__device__ __forceinline__ H4 split4(float4 x) {
    H4 r;
    r.h[0] = (_Float16)x.x; r.l[0] = (_Float16)(x.x - (float)r.h[0]);
    r.h[1] = (_Float16)x.y; r.l[1] = (_Float16)(x.y - (float)r.h[1]);
    r.h[2] = (_Float16)x.z; r.l[2] = (_Float16)(x.z - (float)r.h[2]);
    r.h[3] = (_Float16)x.w; r.l[3] = (_Float16)(x.w - (float)r.h[3]);
    return r;
}

// Build a v16h fragment from two contiguous 8-half runs at p and p+step.
__device__ __forceinline__ v16h ldsfrag(const _Float16* p, int step) {
    v8h p0 = *(const v8h*)(p);
    v8h p1 = *(const v8h*)(p + step);
    return __builtin_shufflevector(p0, p1, 0,1,2,3,4,5,6,7,8,9,10,11,12,13,14,15);
}

#define WMMA_F16(A, B, C) \
    __builtin_amdgcn_wmma_f32_16x16x32_f16(false, (A), false, (B), (short)0, (C), false, false)

// ---------------------------------------------------------------------------
// Reciprocal L2 row norms for Q (blockIdx.y==0) and K (blockIdx.y==1)
// ---------------------------------------------------------------------------
__global__ __launch_bounds__(256)
void rnorm_kernel(const float* __restrict__ q, const float* __restrict__ k,
                  float* __restrict__ rq, float* __restrict__ rk) {
    const int row = blockIdx.x;
    const float* src = (blockIdx.y == 0) ? q : k;
    float* dst = (blockIdx.y == 0) ? rq : rk;
    const float4 v = ((const float4*)(src + (size_t)row * DIM))[threadIdx.x];
    float ss = v.x * v.x + v.y * v.y + v.z * v.z + v.w * v.w;
    __shared__ float red[256];
    red[threadIdx.x] = ss;
    __syncthreads();
    for (int s = 128; s > 0; s >>= 1) {
        if (threadIdx.x < s) red[threadIdx.x] += red[threadIdx.x + s];
        __syncthreads();
    }
    if (threadIdx.x == 0) dst[row] = 1.0f / fmaxf(sqrtf(red[0]), EPSN);
}

// ---------------------------------------------------------------------------
// GEMM1: attn[k,q] = (K_k . Q_q) * rk[k] * rq[q]       attn is [NK][NQ]
// Split-f16 WMMA: Ahi*Bhi + Alo*Bhi + Ahi*Blo (f32 accumulate).
// Block 128x128, 8 waves (2M x 4N), wave = 64x32 = 4x2 WMMA tiles.
// ---------------------------------------------------------------------------
__global__ __launch_bounds__(256)
void gemm1_kernel(const float* __restrict__ Kmat, const float* __restrict__ Qmat,
                  const float* __restrict__ rk, const float* __restrict__ rq,
                  float* __restrict__ attn) {
    __shared__ _Float16 Ahi[128 * STR];
    __shared__ _Float16 Alo[128 * STR];
    __shared__ _Float16 Bhi[128 * STR];
    __shared__ _Float16 Blo[128 * STR];

    const int tid   = threadIdx.x;
    const int lane  = tid & 31;
    const int waveM = (tid >> 5) & 1;
    const int waveN = (tid >> 5) >> 1;
    const int mBase = blockIdx.y * 128;  // key rows
    const int nBase = blockIdx.x * 128;  // query rows (attn cols)
    const int lrow  = lane & 15;
    const int half  = lane >> 4;

    v8f acc[4][2];
    #pragma unroll
    for (int i = 0; i < 4; ++i)
        #pragma unroll
        for (int j = 0; j < 2; ++j)
            acc[i][j] = (v8f){0.f,0.f,0.f,0.f,0.f,0.f,0.f,0.f};

    const int grow = tid >> 1;           // staged row (0..127)
    const int gcb  = (tid & 1) * 16;     // 16 floats per thread per matrix
    const float* Krow = Kmat + (size_t)(mBase + grow) * DIM + gcb;
    const float* Qrow = Qmat + (size_t)(nBase + grow) * DIM + gcb;

    for (int k0 = 0; k0 < DIM; k0 += BK1) {
        float4 ka[4], qa[4];
        #pragma unroll
        for (int u = 0; u < 4; ++u) {
            ka[u] = *(const float4*)(Krow + k0 + u * 4);
            qa[u] = *(const float4*)(Qrow + k0 + u * 4);
        }
        __syncthreads();                 // previous iteration's LDS reads done
        #pragma unroll
        for (int u = 0; u < 4; ++u) {
            const H4 a = split4(ka[u]);
            *(v4h*)(&Ahi[grow * STR + gcb + u * 4]) = a.h;
            *(v4h*)(&Alo[grow * STR + gcb + u * 4]) = a.l;
            const H4 b = split4(qa[u]);
            *(v4h*)(&Bhi[grow * STR + gcb + u * 4]) = b.h;
            *(v4h*)(&Blo[grow * STR + gcb + u * 4]) = b.l;
        }
        __syncthreads();

        // B fragments: 16 contiguous K at half*16 (same k order as A)
        v16h bh[2], bl[2];
        #pragma unroll
        for (int j = 0; j < 2; ++j) {
            const int off = (waveN * 32 + j * 16 + lrow) * STR + half * 16;
            bh[j] = ldsfrag(&Bhi[off], 8);
            bl[j] = ldsfrag(&Blo[off], 8);
        }
        // A fragments: 16 contiguous K at half*16 (matches B)
        #pragma unroll
        for (int i = 0; i < 4; ++i) {
            const int off = (waveM * 64 + i * 16 + lrow) * STR + half * 16;
            const v16h ah = ldsfrag(&Ahi[off], 8);
            const v16h al = ldsfrag(&Alo[off], 8);
            #pragma unroll
            for (int j = 0; j < 2; ++j) {
                acc[i][j] = WMMA_F16(ah, bh[j], acc[i][j]);
                acc[i][j] = WMMA_F16(al, bh[j], acc[i][j]);
                acc[i][j] = WMMA_F16(ah, bl[j], acc[i][j]);
            }
        }
    }

    // epilogue: exact cosine scaling in f32
    #pragma unroll
    for (int i = 0; i < 4; ++i) {
        #pragma unroll
        for (int j = 0; j < 2; ++j) {
            const int col = nBase + waveN * 32 + j * 16 + lrow;
            const float rqv = rq[col];
            #pragma unroll
            for (int r = 0; r < 8; ++r) {
                const int mrow = mBase + waveM * 64 + i * 16 + r + half * 8;
                attn[(size_t)mrow * NQ + col] = acc[i][j][r] * rk[mrow] * rqv;
            }
        }
    }
}

// ---------------------------------------------------------------------------
// GEMM2: weighted[q,d] = sum_k attn[k,q] * V[k,d]      weighted is [NQ][DIM]
// attn/V are k-major: stage f32 tiles, build transposed hi/lo fragments
// with strided b32 LDS reads + register-side f16 split.
// ---------------------------------------------------------------------------
__global__ __launch_bounds__(256)
void gemm2_kernel(const float* __restrict__ attn, const float* __restrict__ Vmat,
                  float* __restrict__ Wout) {
    __shared__ float At[BK2 * LDT];   // [k][q]
    __shared__ float Vs[BK2 * LDT];   // [k][d]

    const int tid   = threadIdx.x;
    const int lane  = tid & 31;
    const int waveM = (tid >> 5) & 1;
    const int waveN = (tid >> 5) >> 1;
    const int qBase = blockIdx.y * 128;  // M
    const int dBase = blockIdx.x * 128;  // N
    const int lrow  = lane & 15;
    const int half  = lane >> 4;
    const int sA    = half * 16;         // A K-run start
    const int sB    = half * 16;         // B K-run start

    v8f acc[4][2];
    #pragma unroll
    for (int i = 0; i < 4; ++i)
        #pragma unroll
        for (int j = 0; j < 2; ++j)
            acc[i][j] = (v8f){0.f,0.f,0.f,0.f,0.f,0.f,0.f,0.f};

    for (int k0 = 0; k0 < NK; k0 += BK2) {
        float4 av[4], vv[4];
        #pragma unroll
        for (int u = 0; u < 4; ++u) {   // 1024 float4 per tile, 4 per thread
            const int idx = tid + u * 256;
            const int kr  = idx >> 5;        // 0..31
            const int c   = (idx & 31) * 4;  // 0..124
            av[u] = *(const float4*)(attn + (size_t)(k0 + kr) * NQ + qBase + c);
            vv[u] = *(const float4*)(Vmat + (size_t)(k0 + kr) * DIM + dBase + c);
        }
        __syncthreads();
        #pragma unroll
        for (int u = 0; u < 4; ++u) {
            const int idx = tid + u * 256;
            const int kr  = idx >> 5;
            const int c   = (idx & 31) * 4;
            *(float4*)(&At[kr * LDT + c]) = av[u];
            *(float4*)(&Vs[kr * LDT + c]) = vv[u];
        }
        __syncthreads();

        // B fragments from V (fixed d, K contiguous per lane layout)
        v16h bh[2], bl[2];
        #pragma unroll
        for (int j = 0; j < 2; ++j) {
            const int dc = waveN * 32 + j * 16 + lrow;
            #pragma unroll
            for (int r = 0; r < 16; ++r) {
                const float x = Vs[(sB + r) * LDT + dc];
                const _Float16 h = (_Float16)x;
                bh[j][r] = h;
                bl[j][r] = (_Float16)(x - (float)h);
            }
        }
        // A fragments from attn^T (fixed q, 16 contiguous K at sA)
        #pragma unroll
        for (int i = 0; i < 4; ++i) {
            const int qr = waveM * 64 + i * 16 + lrow;
            v16h ah, al;
            #pragma unroll
            for (int r = 0; r < 8; ++r) {
                const float x0 = At[(sA + r) * LDT + qr];
                const float x1 = At[(sA + 8 + r) * LDT + qr];
                const _Float16 h0 = (_Float16)x0;
                const _Float16 h1 = (_Float16)x1;
                ah[r]     = h0;  al[r]     = (_Float16)(x0 - (float)h0);
                ah[8 + r] = h1;  al[8 + r] = (_Float16)(x1 - (float)h1);
            }
            #pragma unroll
            for (int j = 0; j < 2; ++j) {
                acc[i][j] = WMMA_F16(ah, bh[j], acc[i][j]);
                acc[i][j] = WMMA_F16(al, bh[j], acc[i][j]);
                acc[i][j] = WMMA_F16(ah, bl[j], acc[i][j]);
            }
        }
    }

    #pragma unroll
    for (int i = 0; i < 4; ++i) {
        #pragma unroll
        for (int j = 0; j < 2; ++j) {
            const int dcol = dBase + waveN * 32 + j * 16 + lrow;
            #pragma unroll
            for (int r = 0; r < 8; ++r) {
                const int qrow = qBase + waveM * 64 + i * 16 + r + half * 8;
                Wout[(size_t)qrow * DIM + dcol] = acc[i][j][r];
            }
        }
    }
}

// ---------------------------------------------------------------------------
extern "C" void kernel_launch(void* const* d_in, const int* in_sizes, int n_in,
                              void* d_out, int out_size, void* d_ws, size_t ws_size,
                              hipStream_t stream) {
    const float* q = (const float*)d_in[0];
    const float* k = (const float*)d_in[1];
    const float* v = (const float*)d_in[2];

    float* out      = (float*)d_out;
    float* weighted = out;                       // [NQ][DIM]
    float* attn     = out + (size_t)NQ * DIM;    // [NK][NQ]

    float* rq = (float*)d_ws;                    // [NQ]
    float* rk = rq + NQ;                         // [NK]

    rnorm_kernel<<<dim3(NQ, 2), 256, 0, stream>>>(q, k, rq, rk);
    gemm1_kernel<<<dim3(NQ / 128, NK / 128), 256, 0, stream>>>(k, q, rk, rq, attn);
    gemm2_kernel<<<dim3(DIM / 128, NQ / 128), 256, 0, stream>>>(attn, v, weighted);
}
